// EncoderLayers_50998441673230
// MI455X (gfx1250) — matrix operand order repaired and checked by reading, not compiled, not run on hardware
//
#include <hip/hip_runtime.h>
#include <math.h>

// ---------------------------------------------------------------------------
// MI455X (gfx1250) transformer encoder layer, round 4.
// - v_wmma_f32_16x16x32_bf16 for all matrix math (f32 accumulate)
// - LDS ping-pong + register staging: HBM loads of tile k+1 overlap WMMA of k
// - Mask tile fetched via Tensor Data Mover (tensor_load_to_lds + TENSORcnt)
//   using the 6-arg builtin form (clang-23 / therock-10.0 toolchain).
// ---------------------------------------------------------------------------

typedef __attribute__((ext_vector_type(16))) __bf16 v16bf;
typedef __attribute__((ext_vector_type(8)))  float  v8f;
typedef __attribute__((ext_vector_type(4)))  unsigned int v4u;
typedef __attribute__((ext_vector_type(8)))  int v8i_t;
typedef __attribute__((ext_vector_type(4)))  int v4i_t;

union BF16Frag { v16bf v; unsigned int u[8]; uint4 u4[2]; };

__device__ __forceinline__ unsigned short f2bf(float f) {
  union { float f; unsigned int u; } c; c.f = f;
  unsigned int r = c.u + 0x7FFFu + ((c.u >> 16) & 1u);   // round-to-nearest-even
  return (unsigned short)(r >> 16);
}
__device__ __forceinline__ unsigned int pack2(float lo, float hi) {
  return (unsigned int)f2bf(lo) | ((unsigned int)f2bf(hi) << 16);
}

#if defined(__has_builtin)
#if __has_builtin(__builtin_amdgcn_tensor_load_to_lds) && \
    __has_builtin(__builtin_amdgcn_s_wait_tensorcnt)
#define HAVE_TDM 1
#endif
#endif
#ifndef HAVE_TDM
#define HAVE_TDM 0
#endif

#if HAVE_TDM
// TDM load of a 64x64 f32 tile from a 2048x2048 row-major tensor into LDS.
// D# per CDNA5 ISA ch8: group0 = {count|flags, lds_addr, gaddr_lo, gaddr_hi|type},
// group1 = {data_size, tensor_dim0/1=2048, tile 64x64, dim0_stride=2048}.
// Groups 2/3 (and the extra dwords of the 6-arg builtin) are zero: 2-D tensor.
__device__ __forceinline__ void tdm_load_mask64(unsigned lds_off, const float* g) {
  unsigned long long ga = (unsigned long long)(uintptr_t)g;
  v4u g0;
  g0[0] = 1u;                                          // count=1, user D#
  g0[1] = lds_off;                                     // LDS byte address
  g0[2] = (unsigned)(ga & 0xFFFFFFFFu);                // global_addr[31:0]
  g0[3] = (unsigned)((ga >> 32) & 0x01FFFFFFu)         // global_addr[56:32]
          | 0x80000000u;                               // type=2 -> bits[127:126]=10b
  v8i_t g1;
  g1[0] = 0x00020000;                                  // data_size=4B, wg_mask=0
  g1[1] = (int)(2048u << 16);                          // tensor_dim0=2048 (lo16)
  g1[2] = (int)(2048u << 16);                          // td0 hi=0 | tensor_dim1 lo16
  g1[3] = (int)(64u << 16);                            // td1 hi=0 | tile_dim0=64
  g1[4] = 64;                                          // tile_dim1=64, tile_dim2=0
  g1[5] = 2048;                                        // tensor_dim0_stride lo32
  g1[6] = 0;                                           // stride hi | dim1_stride lo
  g1[7] = 0;
  v4i_t z4 = {0, 0, 0, 0};
  v8i_t z8 = {0, 0, 0, 0, 0, 0, 0, 0};
  __builtin_amdgcn_tensor_load_to_lds(g0, g1, z4, z4, z8, 0);
}
#endif

// ---------------------------------------------------------------------------
// GEMM: C[M,N] = A[M,K] @ B[K,N] + bias (+ residual) (+ ReLU)
// Block 64x256, 8 waves (2x4), wave tile 32x64 (2x4 frags), BK=32.
// LDS ping-pong; A-frag = 2x ds_load_b128, B-frag (pair-packed) = 8x b32.
// ---------------------------------------------------------------------------
template<bool RELU, bool HAS_RES>
__global__ __launch_bounds__(256)
void gemm_bias(const float* __restrict__ A, const float* __restrict__ Bw,
               const float* __restrict__ bias, const float* __restrict__ res,
               float* __restrict__ C, int N, int K)
{
  constexpr int BM = 64, BN = 256, BK = 32;
  constexpr int SA  = 40;        // ushorts; 80 B rows (16B aligned)
  constexpr int SB2 = BN + 4;    // uints; 1040 B rows (16B aligned)
  __shared__ unsigned short As[2][BM * SA];
  __shared__ unsigned int   Bs2[2][(BK / 2) * SB2];

  const int tid  = threadIdx.x;
  const int wave = tid >> 5, lane = tid & 31;
  const int wm = wave >> 2, wn = wave & 3;
  const int lm = lane & 15, lh = lane >> 4;
  const int row0 = blockIdx.y * BM;
  const int col0 = blockIdx.x * BN;

  float4 ra[2], rb[4][2];
  auto stage_regs = [&](int k0) {
#pragma unroll
    for (int t = 0; t < 2; ++t) {            // A tile 64x32: 2 float4/thread
      int idx = tid + t * 256;
      int m = idx >> 3, k = (idx & 7) * 4;
      ra[t] = *(const float4*)&A[(size_t)(row0 + m) * K + k0 + k];
    }
#pragma unroll
    for (int t = 0; t < 4; ++t) {            // B tile 32x256: 8 float4/thread
      int idx = tid + t * 256;
      int kp2 = idx >> 6, n = (idx & 63) * 4;
      rb[t][0] = *(const float4*)&Bw[(size_t)(k0 + 2 * kp2)     * N + col0 + n];
      rb[t][1] = *(const float4*)&Bw[(size_t)(k0 + 2 * kp2 + 1) * N + col0 + n];
    }
  };
  auto stage_lds = [&](int buf) {
#pragma unroll
    for (int t = 0; t < 2; ++t) {
      int idx = tid + t * 256;
      int m = idx >> 3, k = (idx & 7) * 4;
      *(uint2*)&As[buf][m * SA + k] =
          make_uint2(pack2(ra[t].x, ra[t].y), pack2(ra[t].z, ra[t].w));
    }
#pragma unroll
    for (int t = 0; t < 4; ++t) {
      int idx = tid + t * 256;
      int kp2 = idx >> 6, n = (idx & 63) * 4;
      *(uint4*)&Bs2[buf][kp2 * SB2 + n] =
          make_uint4(pack2(rb[t][0].x, rb[t][1].x), pack2(rb[t][0].y, rb[t][1].y),
                     pack2(rb[t][0].z, rb[t][1].z), pack2(rb[t][0].w, rb[t][1].w));
    }
  };

  v8f acc[2][4] = {};
  stage_regs(0);
  stage_lds(0);
  __syncthreads();

  const int nk = K / BK;
  for (int kk = 0; kk < nk; ++kk) {
    const int cur = kk & 1;
    if (kk + 1 < nk) stage_regs((kk + 1) * BK);   // HBM loads overlap compute

    BF16Frag a[2], b[4];
#pragma unroll
    for (int i = 0; i < 2; ++i) {                 // A-frag: quads at +0,+16 B
      int m = wm * 32 + i * 16 + lm;
      const uint4* p = (const uint4*)&As[cur][m * SA + 16 * lh];
      a[i].u4[0] = p[0];
      a[i].u4[1] = p[1];
    }
#pragma unroll
    for (int j = 0; j < 4; ++j) {                 // B-frag: pair-packed rows
      int n = wn * 64 + j * 16 + lm;
#pragma unroll
      for (int v = 0; v < 8; ++v)
        b[j].u[v] = Bs2[cur][(v + 8 * lh) * SB2 + n];
    }
#pragma unroll
    for (int i = 0; i < 2; ++i)
#pragma unroll
      for (int j = 0; j < 4; ++j)
        acc[i][j] = __builtin_amdgcn_wmma_f32_16x16x32_bf16(
            false, a[i].v, false, b[j].v, (short)0, acc[i][j], false, false);

    if (kk + 1 < nk) stage_lds(cur ^ 1);
    __syncthreads();
  }

  // Epilogue: C element (m = r + 8*lh, n = lane&15) per 16x16 f32 D layout.
#pragma unroll
  for (int i = 0; i < 2; ++i) {
#pragma unroll
    for (int j = 0; j < 4; ++j) {
      int col = col0 + wn * 64 + j * 16 + lm;
      float bv = bias[col];
#pragma unroll
      for (int r = 0; r < 8; ++r) {
        int row = row0 + wm * 32 + i * 16 + r + 8 * lh;
        float val = acc[i][j][r] + bv;
        if (HAS_RES) val += res[(size_t)row * N + col];
        if (RELU)    val = fmaxf(val, 0.0f);
        C[(size_t)row * N + col] = val;
      }
    }
  }
}

// ---------------------------------------------------------------------------
// Flash attention: block = (64 q-rows, head h, batch b); 4 waves x 16 q-rows.
// K/V tiles ping-pong with register staging; mask tile via TDM (double-
// buffered LDS, TENSORcnt-fenced). qkv: [b,s,h*192+{0,64,128}+d].
// ---------------------------------------------------------------------------
__global__ __launch_bounds__(128)
void flash_attn(const float* __restrict__ qkv, const float* __restrict__ mask,
                float* __restrict__ ctx)
{
  constexpr int S = 2048, DQKV = 3072, HD = 64, TQ = 64, TK = 64;
  constexpr int SD  = 72;        // ushorts; 144 B rows (16B aligned)
  constexpr int SV2 = HD + 4;    // uints;  272 B rows (16B aligned)
  __shared__ unsigned short Qs[TQ * SD];
  __shared__ unsigned short Ks[2][TK * SD];
  __shared__ unsigned int   Vs2[2][(TK / 2) * SV2];
  __shared__ unsigned short Ps[TQ * SD];
  __shared__ float          Ms[2][TQ * TK];      // mask tile (TDM destination)

  const int tid  = threadIdx.x;
  const int wq   = tid >> 5;
  const int lane = tid & 31;
  const int lm = lane & 15, lh = lane >> 4;
  const int q0 = blockIdx.x * TQ;
  const int h  = blockIdx.y;
  const int b  = blockIdx.z;

  const size_t base = (size_t)b * S * DQKV + (size_t)h * 3 * HD;

  // Q tile -> LDS (once)
#pragma unroll
  for (int t = 0; t < 8; ++t) {
    int idx = tid + t * 128;
    int m = idx >> 4, d = (idx & 15) * 4;
    const float4 f = *(const float4*)&qkv[base + (size_t)(q0 + m) * DQKV + d];
    *(uint2*)&Qs[m * SD + d] = make_uint2(pack2(f.x, f.y), pack2(f.z, f.w));
  }

  float4 rk[8], rva[4], rvb[4];
  auto stage_regs = [&](int kt) {
#pragma unroll
    for (int t = 0; t < 8; ++t) {
      int idx = tid + t * 128;
      int m = idx >> 4, d = (idx & 15) * 4;
      rk[t] = *(const float4*)&qkv[base + (size_t)(kt + m) * DQKV + HD + d];
    }
#pragma unroll
    for (int t = 0; t < 4; ++t) {
      int idx = tid + t * 128;
      int kp2 = idx >> 4, d = (idx & 15) * 4;
      rva[t] = *(const float4*)&qkv[base + (size_t)(kt + 2 * kp2)     * DQKV + 2 * HD + d];
      rvb[t] = *(const float4*)&qkv[base + (size_t)(kt + 2 * kp2 + 1) * DQKV + 2 * HD + d];
    }
  };
  auto stage_lds = [&](int buf) {
#pragma unroll
    for (int t = 0; t < 8; ++t) {
      int idx = tid + t * 128;
      int m = idx >> 4, d = (idx & 15) * 4;
      *(uint2*)&Ks[buf][m * SD + d] =
          make_uint2(pack2(rk[t].x, rk[t].y), pack2(rk[t].z, rk[t].w));
    }
#pragma unroll
    for (int t = 0; t < 4; ++t) {
      int idx = tid + t * 128;
      int kp2 = idx >> 4, d = (idx & 15) * 4;
      *(uint4*)&Vs2[buf][kp2 * SV2 + d] =
          make_uint4(pack2(rva[t].x, rvb[t].x), pack2(rva[t].y, rvb[t].y),
                     pack2(rva[t].z, rvb[t].z), pack2(rva[t].w, rvb[t].w));
    }
  };
  auto load_mask_tile = [&](int buf, int kt) {
#if HAVE_TDM
    if (wq == 0)                                     // TDM ignores EXEC: gate per-wave
      tdm_load_mask64((unsigned)(uintptr_t)&Ms[buf][0],
                      &mask[(size_t)q0 * S + kt]);
#else
#pragma unroll
    for (int t = 0; t < 8; ++t) {                    // fallback: cooperative load
      int idx = tid + t * 128;
      int m = idx >> 4, d = (idx & 15) * 4;
      *(float4*)&Ms[buf][m * TK + d] =
          *(const float4*)&mask[(size_t)(q0 + m) * S + kt + d];
    }
#endif
  };
  auto wait_mask = [&]() {
#if HAVE_TDM
    if (wq == 0) __builtin_amdgcn_s_wait_tensorcnt(0);
#endif
  };

  float mrow[8], lrow[8];
#pragma unroll
  for (int r = 0; r < 8; ++r) { mrow[r] = -1e30f; lrow[r] = 0.0f; }
  v8f o[4] = {};

  // prologue: stage tile 0
  stage_regs(0);
  stage_lds(0);
  load_mask_tile(0, 0);
  wait_mask();
  __syncthreads();

  const int nt = S / TK;
  for (int it = 0; it < nt; ++it) {
    const int cur = it & 1;
    const int kt  = it * TK;
    if (it + 1 < nt) {
      stage_regs(kt + TK);                           // HBM loads overlap compute
      load_mask_tile(cur ^ 1, kt + TK);              // TDM overlaps compute
    }

    // S = Q @ K^T
    v8f s[4] = {};
#pragma unroll
    for (int ks = 0; ks < 2; ++ks) {
      BF16Frag a;
      {
        int m = wq * 16 + lm;
        const uint4* p = (const uint4*)&Qs[m * SD + ks * 32 + 16 * lh];
        a.u4[0] = p[0];
        a.u4[1] = p[1];
      }
#pragma unroll
      for (int j = 0; j < 4; ++j) {
        BF16Frag bb;                                 // K^T: contiguous along dim
        int n = j * 16 + lm;
        const uint4* p = (const uint4*)&Ks[cur][n * SD + ks * 32 + 16 * lh];
        bb.u4[0] = p[0];
        bb.u4[1] = p[1];
        s[j] = __builtin_amdgcn_wmma_f32_16x16x32_bf16(
            false, a.v, false, bb.v, (short)0, s[j], false, false);
      }
    }

    // scale + mask (mask tile in LDS)
    const float scale = 0.125f;                      // 1/sqrt(64)
#pragma unroll
    for (int j = 0; j < 4; ++j)
#pragma unroll
      for (int r = 0; r < 8; ++r) {
        int mrel = wq * 16 + r + 8 * lh;
        s[j][r] = s[j][r] * scale + Ms[cur][mrel * TK + j * 16 + lm];
      }

    // online softmax: row m = r + 8*lh lives in this lane's 16-lane half
#pragma unroll
    for (int r = 0; r < 8; ++r) {
      float mx = fmaxf(fmaxf(s[0][r], s[1][r]), fmaxf(s[2][r], s[3][r]));
#pragma unroll
      for (int off = 8; off >= 1; off >>= 1)
        mx = fmaxf(mx, __shfl_xor(mx, off, 32));
      float nm = fmaxf(mrow[r], mx);
      float corr = __expf(mrow[r] - nm);
      mrow[r] = nm;
      lrow[r] *= corr;
#pragma unroll
      for (int j = 0; j < 4; ++j) o[j][r] = o[j][r] * corr;
      float rs = 0.0f;
#pragma unroll
      for (int j = 0; j < 4; ++j) {
        float p = __expf(s[j][r] - nm);
        s[j][r] = p;
        rs += p;
      }
#pragma unroll
      for (int off = 8; off >= 1; off >>= 1)
        rs += __shfl_xor(rs, off, 32);
      lrow[r] += rs;
    }

    // P (C-layout) -> LDS bf16 -> A-layout reload; rows are wave-private.
#pragma unroll
    for (int j = 0; j < 4; ++j)
#pragma unroll
      for (int r = 0; r < 8; ++r)
        Ps[(wq * 16 + r + 8 * lh) * SD + j * 16 + lm] = f2bf(s[j][r]);

    // O += P @ V
#pragma unroll
    for (int ks = 0; ks < 2; ++ks) {
      BF16Frag a;
      {
        int m = wq * 16 + lm;
        const uint4* p = (const uint4*)&Ps[m * SD + ks * 32 + 16 * lh];
        a.u4[0] = p[0];
        a.u4[1] = p[1];
      }
#pragma unroll
      for (int j = 0; j < 4; ++j) {
        BF16Frag bb;                                 // pair-packed V rows
        int n = j * 16 + lm;
#pragma unroll
        for (int v = 0; v < 8; ++v)
          bb.u[v] = Vs2[cur][(ks * 16 + v + 8 * lh) * SV2 + n];
        o[j] = __builtin_amdgcn_wmma_f32_16x16x32_bf16(
            false, a.v, false, bb.v, (short)0, o[j], false, false);
      }
    }

    if (it + 1 < nt) stage_lds(cur ^ 1);
    wait_mask();                                     // TENSORcnt fence (wave 0)
    __syncthreads();
  }

  // normalize and write ctx
#pragma unroll
  for (int r = 0; r < 8; ++r) {
    float inv = 1.0f / lrow[r];
    int qrow = q0 + wq * 16 + r + 8 * lh;
#pragma unroll
    for (int j = 0; j < 4; ++j) {
      int d = j * 16 + lm;
      ctx[((size_t)b * S + qrow) * 1024 + h * 64 + d] = o[j][r] * inv;
    }
  }
}

// ---------------------------------------------------------------------------
// Row LayerNorm over D=1024; one 256-thread block per row (8 wave32), float4.
// ---------------------------------------------------------------------------
__global__ __launch_bounds__(256)
void layernorm_kernel(const float* __restrict__ in, const float* __restrict__ gamma,
                      const float* __restrict__ beta, float* __restrict__ out)
{
  constexpr int D = 1024;
  __shared__ float red[8];
  const int row = blockIdx.x;
  const int tid = threadIdx.x;
  const int c0  = tid * 4;

  const float4 f = *(const float4*)&in[(size_t)row * D + c0];
  float sum = f.x + f.y + f.z + f.w;
#pragma unroll
  for (int off = 16; off >= 1; off >>= 1) sum += __shfl_xor(sum, off, 32);
  if ((tid & 31) == 0) red[tid >> 5] = sum;
  __syncthreads();
  float tot = 0.0f;
#pragma unroll
  for (int w = 0; w < 8; ++w) tot += red[w];
  float mean = tot * (1.0f / 1024.0f);
  __syncthreads();

  float dx = f.x - mean, dy = f.y - mean, dz = f.z - mean, dw = f.w - mean;
  float vs = dx * dx + dy * dy + dz * dz + dw * dw;
#pragma unroll
  for (int off = 16; off >= 1; off >>= 1) vs += __shfl_xor(vs, off, 32);
  if ((tid & 31) == 0) red[tid >> 5] = vs;
  __syncthreads();
  float vtot = 0.0f;
#pragma unroll
  for (int w = 0; w < 8; ++w) vtot += red[w];
  float inv = rsqrtf(vtot * (1.0f / 1024.0f) + 1e-5f);

  const float4 g  = *(const float4*)&gamma[c0];
  const float4 be = *(const float4*)&beta[c0];
  *(float4*)&out[(size_t)row * D + c0] =
      make_float4(g.x * dx * inv + be.x, g.y * dy * inv + be.y,
                  g.z * dz * inv + be.z, g.w * dw * inv + be.w);
}

// ---------------------------------------------------------------------------
extern "C" void kernel_launch(void* const* d_in, const int* in_sizes, int n_in,
                              void* d_out, int out_size, void* d_ws, size_t ws_size,
                              hipStream_t stream)
{
  const float* x     = (const float*)d_in[0];
  const float* mask  = (const float*)d_in[1];
  const float* Wqkv  = (const float*)d_in[2];
  const float* bqkv  = (const float*)d_in[3];
  const float* Wo    = (const float*)d_in[4];
  const float* bo    = (const float*)d_in[5];
  const float* gamma = (const float*)d_in[6];
  const float* beta  = (const float*)d_in[7];
  const float* W1    = (const float*)d_in[8];
  const float* b1    = (const float*)d_in[9];
  const float* W2    = (const float*)d_in[10];
  const float* b2    = (const float*)d_in[11];
  float* out = (float*)d_out;

  const int BS = 2 * 2048;                      // 4096 tokens
  float* ws  = (float*)d_ws;
  // Workspace layout (96 MB total, with region reuse):
  float* qkv = ws;                              // [0,   48MB)  4096x3072
  float* ctx = ws + (size_t)BS * 3072;          // [48MB,64MB)  4096x1024
  float* h1  = ws;                              // [0,   64MB)  4096x4096 (qkv+ctx dead)
  float* y1  = ws + (size_t)BS * 4096;          // [64MB,80MB)  4096x1024
  float* x2  = y1 + (size_t)BS * 1024;          // [80MB,96MB)  4096x1024
  float* y2  = y1;                              // reuse y1 (dead after LN)

  dim3 blk(256);
  // 1. qkv = x @ Wqkv + bqkv                   (M=4096, N=3072, K=1024)
  gemm_bias<false, false><<<dim3(3072 / 256, BS / 64), blk, 0, stream>>>(
      x, Wqkv, bqkv, nullptr, qkv, 3072, 1024);
  // 2. attention -> ctx
  flash_attn<<<dim3(2048 / 64, 16, 2), dim3(128), 0, stream>>>(qkv, mask, ctx);
  // 3. y1 = ctx @ Wo + bo + x                  (N=1024, K=1024)
  gemm_bias<false, true><<<dim3(1024 / 256, BS / 64), blk, 0, stream>>>(
      ctx, Wo, bo, x, y1, 1024, 1024);
  // 4. x2 = LN(y1)
  layernorm_kernel<<<dim3(BS), blk, 0, stream>>>(y1, gamma, beta, x2);
  // 5. h1 = relu(x2 @ W1 + b1)                 (N=4096, K=1024)
  gemm_bias<true, false><<<dim3(4096 / 256, BS / 64), blk, 0, stream>>>(
      x2, W1, b1, nullptr, h1, 4096, 1024);
  // 6. y2 = h1 @ W2 + b2 + x2                  (N=1024, K=4096)
  gemm_bias<false, true><<<dim3(1024 / 256, BS / 64), blk, 0, stream>>>(
      h1, W2, b2, x2, y2, 1024, 4096);
  // 7. out = LN(y2)
  layernorm_kernel<<<dim3(BS), blk, 0, stream>>>(y2, gamma, beta, out);
}
